// GCN_3l_13288628814527
// MI455X (gfx1250) — matrix operand order repaired and checked by reading, not compiled, not run on hardware
//
#include <hip/hip_runtime.h>

typedef __attribute__((ext_vector_type(16))) __bf16 v16bf;
typedef __attribute__((ext_vector_type(8)))  __bf16 v8bf;
typedef __attribute__((ext_vector_type(4)))  __bf16 v4bf;
typedef __attribute__((ext_vector_type(8)))  float  v8f;

#define DIM 128

// ---------------- utility kernels ----------------

__global__ void k_zero_f32(float* p, size_t n) {
  size_t i = (size_t)blockIdx.x * blockDim.x + threadIdx.x;
  size_t stride = (size_t)gridDim.x * blockDim.x;
  for (; i < n; i += stride) p[i] = 0.0f;
}

__global__ void k_deg_count(const int* __restrict__ dst, float* deg, int nE) {
  int e = blockIdx.x * blockDim.x + threadIdx.x;
  if (e < nE)
    __hip_atomic_fetch_add(&deg[dst[e]], 1.0f, __ATOMIC_RELAXED,
                           __HIP_MEMORY_SCOPE_AGENT);
}

__global__ void k_dinv(const float* __restrict__ deg, float* dinv, int n) {
  int i = blockIdx.x * blockDim.x + threadIdx.x;
  if (i < n) dinv[i] = rsqrtf(deg[i] + 1.0f);  // +1 = self loop
}

__global__ void k_f32_to_bf16(const float* __restrict__ in, __bf16* out, size_t n) {
  size_t i = ((size_t)blockIdx.x * blockDim.x + threadIdx.x) * 4;
  if (i < n) {
    float4 v = *(const float4*)(in + i);
    v4bf o;
    o[0] = (__bf16)v.x; o[1] = (__bf16)v.y; o[2] = (__bf16)v.z; o[3] = (__bf16)v.w;
    *(v4bf*)(out + i) = o;
  }
}

// Pre-swizzle W (row-major [128 x ncols], K = rows) into per-lane WMMA B order:
// out[((nt*4+kt)*32 + lane)*16 + e] = bf16(W[(kt*32 + (lane>>4)*16 + e)*ncols + nt*16 + (lane&15)])
__global__ void k_swz(const float* __restrict__ W, __bf16* out, int ncols, int ntn) {
  int t = blockIdx.x * blockDim.x + threadIdx.x;
  int total = ntn * 4 * 32 * 16;
  if (t >= total) return;
  int e  = t & 15;
  int L  = (t >> 4) & 31;
  int kt = (t >> 9) & 3;
  int nt = t >> 11;
  int row = kt * 32 + (L >> 4) * 16 + e;   // K index (fan_in)
  int col = nt * 16 + (L & 15);            // N index (fan_out)
  float v = (col < ncols) ? W[row * ncols + col] : 0.0f;
  out[t] = (__bf16)v;
}

// ---------------- WMMA GEMM: D[N x 16*ntn] = A[N x 128](bf16) @ Wsw ----------------
// one wave computes one 16x16 f32 tile, K=128 via 4 chained 16x16x32 bf16 WMMAs

__global__ void k_gemm_wmma(const __bf16* __restrict__ A,
                            const __bf16* __restrict__ Wsw,
                            float* __restrict__ D,
                            int ldd, int ntn, int mtiles) {
  int lane = threadIdx.x & 31;
  int wv   = threadIdx.x >> 5;
  int mPer = (blockDim.x >> 5) / ntn;       // ntn in {1, 8}, blockDim = 256
  int nt   = wv % ntn;
  int mt   = blockIdx.x * mPer + wv / ntn;
  if (mt >= mtiles) return;                  // wave-uniform; EXEC all-1s inside

  int hf = lane >> 4;                        // which 16-lane half
  int lm = lane & 15;
  const __bf16* arow = A + (size_t)(mt * 16 + lm) * DIM + hf * 16;
  const __bf16* bp   = Wsw + ((size_t)nt * 4 * 32 + lane) * 16;

  v8f acc = {0.f, 0.f, 0.f, 0.f, 0.f, 0.f, 0.f, 0.f};
#pragma unroll
  for (int kt = 0; kt < 4; ++kt) {
    // A 16x32 bf16 fragment: VGPRs0-3 <- K = kt*32 + hf*16 .. +7 ; VGPRs4-7 <- +8
    v8bf lo = *(const v8bf*)(arow + kt * 32);
    v8bf hi = *(const v8bf*)(arow + kt * 32 + 8);
    v16bf a;
#pragma unroll
    for (int i = 0; i < 8; ++i) { a[i] = lo[i]; a[i + 8] = hi[i]; }
    // B 32x16 bf16 fragment: one contiguous 32B chunk per lane (pre-swizzled)
    v16bf b = *(const v16bf*)(bp + (size_t)kt * 32 * 16);
    acc = __builtin_amdgcn_wmma_f32_16x16x32_bf16(false, a, false, b,
                                                  (short)0, acc, false, false);
  }
  // D layout: VGPR r -> row 8*hf + r, col = lane&15
  int n = nt * 16 + lm;
  size_t base = (size_t)(mt * 16 + hf * 8) * ldd + n;
#pragma unroll
  for (int r = 0; r < 8; ++r) D[base + (size_t)r * ldd] = acc[r];
}

// ---------------- edge scatter: agg[dst] += dinv[src]*dinv[dst] * h[src] ----------------
// one wave per edge, each lane handles 4 consecutive features (coalesced 512B row)

__global__ void k_scatter(const int* __restrict__ src, const int* __restrict__ dst,
                          const float* __restrict__ dinv, const float* __restrict__ h,
                          float* agg, int nE) {
  long long t = (long long)blockIdx.x * blockDim.x + threadIdx.x;
  int e = (int)(t >> 5);
  if (e >= nE) return;
  int c = ((int)t & 31) * 4;
  int s = src[e], d = dst[e];
  float nrm = dinv[s] * dinv[d];
  float4 hv = *(const float4*)(h + (size_t)s * DIM + c);
  float* o = agg + (size_t)d * DIM + c;
  __hip_atomic_fetch_add(o + 0, nrm * hv.x, __ATOMIC_RELAXED, __HIP_MEMORY_SCOPE_AGENT);
  __hip_atomic_fetch_add(o + 1, nrm * hv.y, __ATOMIC_RELAXED, __HIP_MEMORY_SCOPE_AGENT);
  __hip_atomic_fetch_add(o + 2, nrm * hv.z, __ATOMIC_RELAXED, __HIP_MEMORY_SCOPE_AGENT);
  __hip_atomic_fetch_add(o + 3, nrm * hv.w, __ATOMIC_RELAXED, __HIP_MEMORY_SCOPE_AGENT);
}

// ---------------- epilogues ----------------
// conv: out_bf16 = relu(agg + dinv[i]^2 * h + bias)   (self-loop folded in)

__global__ void k_epi_conv(const float* __restrict__ agg, const float* __restrict__ h,
                           const float* __restrict__ dinv, const float* __restrict__ bias,
                           __bf16* out, int n) {
  long long t = (long long)blockIdx.x * blockDim.x + threadIdx.x;
  int i = (int)(t >> 5);
  if (i >= n) return;
  int c = ((int)t & 31) * 4;
  float sl = dinv[i]; sl *= sl;
  float4 a  = *(const float4*)(agg + (size_t)i * DIM + c);
  float4 hv = *(const float4*)(h + (size_t)i * DIM + c);
  float4 b  = *(const float4*)(bias + c);
  v4bf o;
  o[0] = (__bf16)fmaxf(a.x + sl * hv.x + b.x, 0.f);
  o[1] = (__bf16)fmaxf(a.y + sl * hv.y + b.y, 0.f);
  o[2] = (__bf16)fmaxf(a.z + sl * hv.z + b.z, 0.f);
  o[3] = (__bf16)fmaxf(a.w + sl * hv.w + b.w, 0.f);
  *(v4bf*)(out + (size_t)i * DIM + c) = o;
}

__global__ void k_epi_ffn(const float* __restrict__ h, const float* __restrict__ bias,
                          __bf16* out, int n) {
  long long t = (long long)blockIdx.x * blockDim.x + threadIdx.x;
  int i = (int)(t >> 5);
  if (i >= n) return;
  int c = ((int)t & 31) * 4;
  float4 hv = *(const float4*)(h + (size_t)i * DIM + c);
  float4 b  = *(const float4*)(bias + c);
  v4bf o;
  o[0] = (__bf16)fmaxf(hv.x + b.x, 0.f);
  o[1] = (__bf16)fmaxf(hv.y + b.y, 0.f);
  o[2] = (__bf16)fmaxf(hv.z + b.z, 0.f);
  o[3] = (__bf16)fmaxf(hv.w + b.w, 0.f);
  *(v4bf*)(out + (size_t)i * DIM + c) = o;
}

// final: d_out[i*10+c] = g[i*16+c] + fb2[c]
__global__ void k_out(const float* __restrict__ g, const float* __restrict__ fb2,
                      float* out, int n) {
  int t = blockIdx.x * blockDim.x + threadIdx.x;
  if (t >= n * 10) return;
  int i = t / 10, c = t - i * 10;
  out[t] = g[(size_t)i * 16 + c] + fb2[c];
}

// ---------------- launch ----------------

extern "C" void kernel_launch(void* const* d_in, const int* in_sizes, int n_in,
                              void* d_out, int out_size, void* d_ws, size_t ws_size,
                              hipStream_t stream) {
  const float* x  = (const float*)d_in[0];
  const int* ei   = (const int*)d_in[1];
  const float* W1 = (const float*)d_in[2];  const float* b1  = (const float*)d_in[3];
  const float* W2 = (const float*)d_in[4];  const float* b2  = (const float*)d_in[5];
  const float* W3 = (const float*)d_in[6];  const float* b3  = (const float*)d_in[7];
  const float* fW1= (const float*)d_in[8];  const float* fb1 = (const float*)d_in[9];
  const float* fW2= (const float*)d_in[10]; const float* fb2 = (const float*)d_in[11];

  const int N = in_sizes[0] / DIM;
  const int E = in_sizes[1] / 2;
  const int* srcI = ei;
  const int* dstI = ei + E;

  char* ws = (char*)d_ws;
  size_t off = 0;
  auto alloc = [&](size_t bytes) -> char* {
    char* p = ws + off;
    off += (bytes + 255) & ~(size_t)255;
    return p;
  };
  float*  deg  = (float*)alloc((size_t)N * 4);
  float*  dinv = (float*)alloc((size_t)N * 4);
  __bf16* hbf  = (__bf16*)alloc((size_t)N * DIM * 2);
  float*  hg   = (float*)alloc((size_t)N * DIM * 4);
  float*  agg  = (float*)alloc((size_t)N * DIM * 4);
  __bf16* w1s  = (__bf16*)alloc(8 * 4 * 32 * 16 * 2);
  __bf16* w2s  = (__bf16*)alloc(8 * 4 * 32 * 16 * 2);
  __bf16* w3s  = (__bf16*)alloc(8 * 4 * 32 * 16 * 2);
  __bf16* f1s  = (__bf16*)alloc(8 * 4 * 32 * 16 * 2);
  __bf16* f2s  = (__bf16*)alloc(1 * 4 * 32 * 16 * 2);

  const int BT = 256;
  const int mtiles = N / 16;

  // degrees + normalization
  k_zero_f32<<<1024, BT, 0, stream>>>(deg, (size_t)N);
  k_deg_count<<<(E + BT - 1) / BT, BT, 0, stream>>>(dstI, deg, E);
  k_dinv<<<(N + BT - 1) / BT, BT, 0, stream>>>(deg, dinv, N);

  // weight pre-swizzle into WMMA-B lane order
  int swzT = 8 * 4 * 32 * 16;
  k_swz<<<(swzT + BT - 1) / BT, BT, 0, stream>>>(W1, w1s, DIM, 8);
  k_swz<<<(swzT + BT - 1) / BT, BT, 0, stream>>>(W2, w2s, DIM, 8);
  k_swz<<<(swzT + BT - 1) / BT, BT, 0, stream>>>(W3, w3s, DIM, 8);
  k_swz<<<(swzT + BT - 1) / BT, BT, 0, stream>>>(fW1, f1s, DIM, 8);
  k_swz<<<(4 * 32 * 16 + BT - 1) / BT, BT, 0, stream>>>(fW2, f2s, 10, 1);

  // x -> bf16 features
  size_t nfeat = (size_t)N * DIM;
  k_f32_to_bf16<<<(unsigned)((nfeat / 4 + BT - 1) / BT), BT, 0, stream>>>(x, hbf, nfeat);

  const float* biases[3] = {b1, b2, b3};
  __bf16* wsw[3] = {w1s, w2s, w3s};
  long long scT = (long long)E * 32;
  long long epT = (long long)N * 32;

  for (int l = 0; l < 3; ++l) {
    k_gemm_wmma<<<mtiles, 256, 0, stream>>>(hbf, wsw[l], hg, DIM, 8, mtiles);
    k_zero_f32<<<4096, BT, 0, stream>>>(agg, nfeat);
    k_scatter<<<(unsigned)((scT + BT - 1) / BT), BT, 0, stream>>>(srcI, dstI, dinv, hg, agg, E);
    k_epi_conv<<<(unsigned)((epT + BT - 1) / BT), BT, 0, stream>>>(agg, hg, dinv, biases[l], hbf, N);
  }

  // FFN head
  k_gemm_wmma<<<mtiles, 256, 0, stream>>>(hbf, f1s, hg, DIM, 8, mtiles);
  k_epi_ffn<<<(unsigned)((epT + BT - 1) / BT), BT, 0, stream>>>(hg, fb1, hbf, N);
  k_gemm_wmma<<<(mtiles + 7) / 8, 256, 0, stream>>>(hbf, f2s, hg, 16, 1, mtiles);
  k_out<<<(N * 10 + BT - 1) / BT, BT, 0, stream>>>(hg, fb2, (float*)d_out, N);
}
